// SimtaPlus_28793460752553
// MI455X (gfx1250) — matrix operand order repaired and checked by reading, not compiled, not run on hardware
//
#include <hip/hip_runtime.h>
#include <hip/hip_bf16.h>

typedef _Float16 v8h  __attribute__((ext_vector_type(8)));
typedef _Float16 v16h __attribute__((ext_vector_type(16)));
typedef float    v8f  __attribute__((ext_vector_type(8)));

#define NB   8
#define NT   512
#define NH   64
#define NS   32
#define NATT 32

static constexpr float kBnInv = 0.9999950000374997f;  // 1/sqrt(1+1e-5)

__device__ __forceinline__ float lrelu(float x) { return x > 0.f ? x : 0.01f * x; }

__device__ __forceinline__ v16h cat8(v8h lo, v8h hi) {
  return __builtin_shufflevector(lo, hi, 0,1,2,3,4,5,6,7,8,9,10,11,12,13,14,15);
}

__device__ __forceinline__ v8f wmma_f16(v16h a, v16h b, v8f c) {
  // (neg_a, A, neg_b, B, c_mod, C, reuse_a, reuse_b)
  return __builtin_amdgcn_wmma_f32_16x16x32_f16(false, a, false, b, (short)0, c, false, false);
}

// ---------------------------------------------------------------------------
// k_prep: dt.max scalar (t sorted ascending per row) + x_s chain (s0, s1, s2)
// ---------------------------------------------------------------------------
__global__ __launch_bounds__(256) void k_prep(
    const float* __restrict__ t, const float* __restrict__ x_stat,
    const float* __restrict__ sw, const float* __restrict__ sb,
    const float* __restrict__ sg, const float* __restrict__ sbeta,
    const float* __restrict__ lw_s, const float* __restrict__ lb_s,
    const float* __restrict__ lg_s, const float* __restrict__ lbeta_s,
    float* __restrict__ dtmax, float* __restrict__ xs012)
{
  __shared__ float sbuf[NB * NS];
  const int tid = threadIdx.x;
  if (tid == 0) {
    float m = -1e30f;
    for (int b = 0; b < NB; b++) m = fmaxf(m, t[b * NT + NT - 1] - t[b * NT]);
    dtmax[0] = m + 1e-8f;
  }
  const int b = tid >> 5, c = tid & 31;
  // s0 = lrelu(bn(x_stat @ sw + sb))
  float acc = sb[c];
  for (int k = 0; k < 16; k++) acc += x_stat[b * 16 + k] * sw[k * NS + c];
  acc = lrelu(acc * (sg[c] * kBnInv) + sbeta[c]);
  xs012[0 * NB * NS + tid] = acc;
  sbuf[tid] = acc;
  __syncthreads();
  // s1 = layer-0 stat MLP(s0)
  acc = lb_s[c];
  for (int k = 0; k < NS; k++) acc += sbuf[b * NS + k] * lw_s[k * NS + c];
  acc = lrelu(acc * (lg_s[c] * kBnInv) + lbeta_s[c]);
  __syncthreads();
  xs012[1 * NB * NS + tid] = acc;
  sbuf[tid] = acc;
  __syncthreads();
  // s2 = layer-1 stat MLP(s1)
  acc = lb_s[NS + c];
  for (int k = 0; k < NS; k++) acc += sbuf[b * NS + k] * lw_s[NS * NS + k * NS + c];
  acc = lrelu(acc * (lg_s[NS + c] * kBnInv) + lbeta_s[NS + c]);
  xs012[2 * NB * NS + tid] = acc;
}

// ---------------------------------------------------------------------------
// k_xt0: x_t0 = lrelu(bn(x_temp @ tw + tb))   (unmasked, matches reference)
// ---------------------------------------------------------------------------
__global__ __launch_bounds__(64) void k_xt0(
    const float* __restrict__ x_temp, const float* __restrict__ tw,
    const float* __restrict__ tb, const float* __restrict__ tg,
    const float* __restrict__ tbeta, float* __restrict__ xt)
{
  __shared__ float xin[32];
  const int row = blockIdx.x;  // b*NT + i
  const int c = threadIdx.x;
  if (c < 32) xin[c] = x_temp[(size_t)row * 32 + c];
  __syncthreads();
  float acc = tb[c];
#pragma unroll
  for (int k = 0; k < 32; k++) acc += xin[k] * tw[k * NH + c];
  acc = lrelu(acc * (tg[c] * kBnInv) + tbeta[c]);
  xt[(size_t)row * NH + c] = acc;
}

// ---------------------------------------------------------------------------
// k_temp_attn: fused dt-MLP + masked softmax per (b, i) row
// ---------------------------------------------------------------------------
__global__ __launch_bounds__(256) void k_temp_attn(
    const float* __restrict__ t, const int* __restrict__ lengths,
    const float* __restrict__ aw0, const float* __restrict__ ab0,
    const float* __restrict__ ag, const float* __restrict__ abeta,
    const float* __restrict__ aw1, const float* __restrict__ ab1,
    const float* __restrict__ dtmax, float* __restrict__ temp_attn)
{
  __shared__ float w0[NATT], b0[NATT], gg[NATT], be[NATT], w1[NATT];
  __shared__ float red[256];
  const int b = blockIdx.x / NT, i = blockIdx.x % NT;
  const int tid = threadIdx.x;
  if (tid < NATT) {
    w0[tid] = aw0[tid]; b0[tid] = ab0[tid];
    gg[tid] = ag[tid] * kBnInv; be[tid] = abeta[tid]; w1[tid] = aw1[tid];
  }
  __syncthreads();
  const float* trow = t + (size_t)b * NT;
  const float ti = trow[i];
  const float inv = 1.0f / dtmax[0];
  const int len = lengths[b];
  const float bias = ab1[0];
  float aval[2];
#pragma unroll
  for (int q = 0; q < 2; q++) {
    const int j = tid + q * 256;
    const float dt = (trow[j] - ti) * inv;
    float acc = bias;
#pragma unroll
    for (int k = 0; k < NATT; k++) {
      float h = dt * w0[k] + b0[k];
      h = h * gg[k] + be[k];
      h = h > 0.f ? h : 0.01f * h;
      acc += h * w1[k];
    }
    aval[q] = (j < len) ? acc : -9.0e8f;
  }
  // softmax over 512
  red[tid] = fmaxf(aval[0], aval[1]);
  __syncthreads();
  for (int s = 128; s > 0; s >>= 1) {
    if (tid < s) red[tid] = fmaxf(red[tid], red[tid + s]);
    __syncthreads();
  }
  const float rmax = red[0];
  __syncthreads();
  const float e0 = __expf(aval[0] - rmax), e1 = __expf(aval[1] - rmax);
  red[tid] = e0 + e1;
  __syncthreads();
  for (int s = 128; s > 0; s >>= 1) {
    if (tid < s) red[tid] += red[tid + s];
    __syncthreads();
  }
  const float invsum = 1.0f / red[0];
  float* out = temp_attn + ((size_t)b * NT + i) * NT;
  out[tid] = e0 * invsum;
  out[tid + 256] = e1 * invsum;
}

// ---------------------------------------------------------------------------
// k_layer_a: x = lrelu(bn(concat(x_t, x_s) @ W + b)) * mask
// emits f16 row-major xh and channel-major xhT for WMMA fragment loads
// ---------------------------------------------------------------------------
__global__ __launch_bounds__(64) void k_layer_a(
    const float* __restrict__ xt_in, const float* __restrict__ xs,
    const float* __restrict__ W, const float* __restrict__ bia,
    const float* __restrict__ g, const float* __restrict__ beta,
    const int* __restrict__ lengths,
    _Float16* __restrict__ xh, _Float16* __restrict__ xhT)
{
  __shared__ float xin[96];
  const int row = blockIdx.x;
  const int b = row / NT, i = row % NT;
  const int c = threadIdx.x;
  xin[c] = xt_in[(size_t)row * NH + c];
  if (c < NS) xin[NH + c] = xs[b * NS + c];
  __syncthreads();
  float acc = bia[c];
#pragma unroll 8
  for (int k = 0; k < 96; k++) acc += xin[k] * W[k * NH + c];
  acc = lrelu(acc * (g[c] * kBnInv) + beta[c]);
  const float v = (i < lengths[b]) ? acc : 0.f;
  xh[(size_t)row * NH + c] = (_Float16)v;
  xhT[((size_t)b * NH + c) * NT + i] = (_Float16)v;
}

// ---------------------------------------------------------------------------
// k_attn: per 16-query-row tile of one batch:
//   scores = x_i . x_j via WMMA (1/sqrt(d) folded into softmax exp)
//   softmax -> + temp_attn, mask -> f16 probs
//   x_t_out = attn @ x via WMMA, masked
// 4 waves / block; all WMMA issued with full EXEC, uniform trip counts.
// ---------------------------------------------------------------------------
__global__ __launch_bounds__(128) void k_attn(
    const _Float16* __restrict__ xh,    // [B][T][H] row-major
    const _Float16* __restrict__ xhT,   // [B][H][T] channel-major
    const float* __restrict__ temp_attn,// [B][T][T]
    const int* __restrict__ lengths,
    float* __restrict__ xt_out)         // [B][T][H]
{
  __shared__ float    s[16][NT];        // 32 KB raw scores
  __shared__ _Float16 a16[16][NT];      // 16 KB attn probs (f16)
  __shared__ float    red[16][8];
  __shared__ float    rowmax[16], rowsum[16];

  const int b    = blockIdx.x >> 5;     // 32 row-tiles per batch
  const int it   = blockIdx.x & 31;
  const int i0   = it * 16;
  const int tid  = threadIdx.x;
  const int wave = tid >> 5;
  const int lane = tid & 31;
  const int r    = lane & 15;           // M (A/D row) or N (B/D col) index
  const int kh   = lane >> 4;           // K-half selector

  const float* ta_base = temp_attn + ((size_t)b * NT + i0) * NT;
  // warm L2/L0 for the temp_attn rows we will add after softmax
  __builtin_prefetch(ta_base + (size_t)(tid >> 3) * NT + (tid & 7) * 64, 0, 0);

  const _Float16* xb = xh + (size_t)b * NT * NH;

  // A fragments (16x32 f16): rows i0..i0+15, K = 0..31 and 32..63
  const _Float16* arow = xb + (size_t)(i0 + r) * NH;
  const v16h A0 = cat8(*(const v8h*)(arow + kh * 8),      *(const v8h*)(arow + 16 + kh * 8));
  const v16h A1 = cat8(*(const v8h*)(arow + 32 + kh * 8), *(const v8h*)(arow + 48 + kh * 8));

  // GEMM1: raw scores; wave w owns j-tiles w, w+4, ..., w+28 (uniform 8-trip loop)
#pragma unroll 4
  for (int q = 0; q < 8; q++) {
    const int j0 = (wave + 4 * q) * 16;
    const _Float16* brow = xb + (size_t)(j0 + r) * NH;  // B[k][n] = xh[j0+n][k]
    const v16h B0 = cat8(*(const v8h*)(brow + kh * 8),      *(const v8h*)(brow + 16 + kh * 8));
    const v16h B1 = cat8(*(const v8h*)(brow + 32 + kh * 8), *(const v8h*)(brow + 48 + kh * 8));
    v8f c = {};
    c = wmma_f16(A0, B0, c);
    c = wmma_f16(A1, B1, c);
#pragma unroll
    for (int v = 0; v < 8; v++) s[v + 8 * kh][j0 + r] = c[v];
  }
  __syncthreads();

  // softmax over all 512 j of qk/8 (reference softmaxes unmasked qk);
  // the 1/sqrt(64) scale is applied inside exp, not to the WMMA result.
  const float scale = 0.125f;
  const int row = tid >> 3, sub = tid & 7;
  float m = -1e30f;
  for (int j = sub; j < NT; j += 8) m = fmaxf(m, s[row][j]);
  red[row][sub] = m;
  __syncthreads();
  if (sub == 0) {
    float mm = red[row][0];
    for (int q = 1; q < 8; q++) mm = fmaxf(mm, red[row][q]);
    rowmax[row] = mm;
  }
  __syncthreads();
  const float rm = rowmax[row];
  float sm = 0.f;
  for (int j = sub; j < NT; j += 8) sm += __expf((s[row][j] - rm) * scale);
  red[row][sub] = sm;
  __syncthreads();
  if (sub == 0) {
    float ss = 0.f;
    for (int q = 0; q < 8; q++) ss += red[row][q];
    rowsum[row] = ss;
  }
  __syncthreads();
  const float invs = 1.0f / rowsum[row];
  const int len = lengths[b];
  const float* ta = ta_base + (size_t)row * NT;
  for (int j = sub; j < NT; j += 8) {
    const float self = __expf((s[row][j] - rm) * scale) * invs;
    const float v = (j < len) ? (self + ta[j]) : 0.f;
    a16[row][j] = (_Float16)v;
  }
  __syncthreads();

  // GEMM2: out[16][64] = attn[16][512] @ x[512][64]; wave w owns n-tile w
  const int n0 = wave * 16;
  const _Float16* bT = xhT + ((size_t)b * NH + n0 + r) * NT;  // B[k][n] = xhT[n0+n][k]
  v8f acc = {};
#pragma unroll 4
  for (int kt = 0; kt < 16; kt++) {
    const int k0 = kt * 32;
    const _Float16* aro = &a16[r][k0];
    const v16h Af = cat8(*(const v8h*)(aro + kh * 8),       *(const v8h*)(aro + 16 + kh * 8));
    const v16h Bf = cat8(*(const v8h*)(bT + k0 + kh * 8),  *(const v8h*)(bT + k0 + 16 + kh * 8));
    acc = wmma_f16(Af, Bf, acc);
  }
  float* ob = xt_out + (size_t)b * NT * NH;
#pragma unroll
  for (int v = 0; v < 8; v++) {
    const int i = i0 + v + 8 * kh;
    ob[(size_t)i * NH + n0 + r] = (i < len) ? acc[v] : 0.f;
  }
}

// ---------------------------------------------------------------------------
// k_final: classifier + masked diff + first-max argmax, writes output + output_max
// ---------------------------------------------------------------------------
__global__ __launch_bounds__(256) void k_final(
    const float* __restrict__ xt, const float* __restrict__ xs2,
    const float* __restrict__ cw, const float* __restrict__ cb,
    const int* __restrict__ lengths, float* __restrict__ out)
{
  __shared__ float o0[NT], o1[NT], df[NT];
  __shared__ float xss[NS];
  __shared__ float bv[256];
  __shared__ int   bi[256];
  const int b = blockIdx.x, tid = threadIdx.x;
  if (tid < NS) xss[tid] = xs2[b * NS + tid];
  __syncthreads();
  const int len = lengths[b];
#pragma unroll
  for (int q = 0; q < 2; q++) {
    const int i = tid + q * 256;
    const float* xr = xt + ((size_t)b * NT + i) * NH;
    float a0 = cb[0], a1 = cb[1];
    for (int k = 0; k < NH; k++) { const float v = xr[k];  a0 += v * cw[k * 2]; a1 += v * cw[k * 2 + 1]; }
    for (int k = 0; k < NS; k++) { const float v = xss[k]; a0 += v * cw[(NH + k) * 2]; a1 += v * cw[(NH + k) * 2 + 1]; }
    if (i >= len) { a0 = 0.f; a1 = 0.f; }
    out[((size_t)b * NT + i) * 2]     = a0;
    out[((size_t)b * NT + i) * 2 + 1] = a1;
    o0[i] = a0; o1[i] = a1;
    df[i] = (i < len) ? (a1 - a0) : -1e30f;
  }
  __syncthreads();
  // argmax, first occurrence on ties
  { const float v0 = df[tid], v1 = df[tid + 256];
    if (v1 > v0) { bv[tid] = v1; bi[tid] = tid + 256; } else { bv[tid] = v0; bi[tid] = tid; } }
  __syncthreads();
  for (int s2 = 128; s2 > 0; s2 >>= 1) {
    if (tid < s2) {
      const float vo = bv[tid + s2]; const int io = bi[tid + s2];
      if (vo > bv[tid] || (vo == bv[tid] && io < bi[tid])) { bv[tid] = vo; bi[tid] = io; }
    }
    __syncthreads();
  }
  if (tid == 0) {
    const int idx = bi[0];
    out[NB * NT * 2 + b * 2]     = o0[idx];
    out[NB * NT * 2 + b * 2 + 1] = o1[idx];
  }
}

// ---------------------------------------------------------------------------
extern "C" void kernel_launch(void* const* d_in, const int* in_sizes, int n_in,
                              void* d_out, int out_size, void* d_ws, size_t ws_size,
                              hipStream_t stream)
{
  (void)in_sizes; (void)n_in; (void)out_size; (void)ws_size;
  const float* x_temp  = (const float*)d_in[0];
  const float* x_stat  = (const float*)d_in[1];
  const float* t       = (const float*)d_in[2];
  /* d_in[3] = masks (bool) — lengths (d_in[4]) is equivalent: mask_q[b,i] = i < lengths[b] */
  const int*   lengths = (const int*)d_in[4];
  const float* aw0 = (const float*)d_in[5],  *ab0   = (const float*)d_in[6];
  const float* ag  = (const float*)d_in[7],  *abeta = (const float*)d_in[8];
  const float* aw1 = (const float*)d_in[9],  *ab1   = (const float*)d_in[10];
  const float* tw  = (const float*)d_in[11], *tb    = (const float*)d_in[12];
  const float* tg  = (const float*)d_in[13], *tbeta = (const float*)d_in[14];
  const float* sw  = (const float*)d_in[15], *sb    = (const float*)d_in[16];
  const float* sg  = (const float*)d_in[17], *sbeta = (const float*)d_in[18];
  const float* lw_t    = (const float*)d_in[19], *lb_t    = (const float*)d_in[20];
  const float* lg_t    = (const float*)d_in[21], *lbeta_t = (const float*)d_in[22];
  const float* lw_s    = (const float*)d_in[23], *lb_s    = (const float*)d_in[24];
  const float* lg_s    = (const float*)d_in[25], *lbeta_s = (const float*)d_in[26];
  const float* cw  = (const float*)d_in[27], *cb = (const float*)d_in[28];
  float* out = (float*)d_out;

  // workspace layout (all 64B aligned; ~11 MB total, L2-resident)
  char* ws = (char*)d_ws;
  float*    dtmax = (float*)ws;
  float*    TA    = (float*)(ws + 64);                          // 8 MB temp_attn
  float*    xtA   = (float*)(ws + 64 + (size_t)NB*NT*NT*4);     // 1 MB
  float*    xtB   = xtA + (size_t)NB*NT*NH;                     // 1 MB
  float*    xs012 = xtB + (size_t)NB*NT*NH;                     // 3 KB
  _Float16* xh    = (_Float16*)(xs012 + 3*NB*NS);               // 512 KB
  _Float16* xhT   = xh + (size_t)NB*NT*NH;                      // 512 KB

  k_prep<<<1, 256, 0, stream>>>(t, x_stat, sw, sb, sg, sbeta,
                                lw_s, lb_s, lg_s, lbeta_s, dtmax, xs012);
  k_xt0<<<NB*NT, 64, 0, stream>>>(x_temp, tw, tb, tg, tbeta, xtA);
  k_temp_attn<<<NB*NT, 256, 0, stream>>>(t, lengths, aw0, ab0, ag, abeta,
                                         aw1, ab1, dtmax, TA);
  // layer 0
  k_layer_a<<<NB*NT, 64, 0, stream>>>(xtA, xs012 + 0*NB*NS,
                                      lw_t, lb_t, lg_t, lbeta_t, lengths, xh, xhT);
  k_attn<<<NB*(NT/16), 128, 0, stream>>>(xh, xhT, TA, lengths, xtB);
  // layer 1
  k_layer_a<<<NB*NT, 64, 0, stream>>>(xtB, xs012 + 1*NB*NS,
                                      lw_t + 96*NH, lb_t + NH, lg_t + NH, lbeta_t + NH,
                                      lengths, xh, xhT);
  k_attn<<<NB*(NT/16), 128, 0, stream>>>(xh, xhT, TA, lengths, xtA);
  k_final<<<NB, 256, 0, stream>>>(xtA, xs012 + 2*NB*NS, cw, cb, lengths, out);
}
